// GORU_7799660610191
// MI455X (gfx1250) — matrix operand order repaired and checked by reading, not compiled, not run on hardware
//
#include <hip/hip_runtime.h>
#include <cstdint>

#define BATCH 8192
#define DIN   1024
#define DOUT  1024
#define CAP   10

typedef __attribute__((ext_vector_type(16))) __bf16 v16bf;
typedef __attribute__((ext_vector_type(8)))  float  v8f;
typedef __attribute__((ext_vector_type(4)))  unsigned int u32x4;
typedef __attribute__((ext_vector_type(8)))  int  i32x8;
typedef __attribute__((ext_vector_type(4)))  int  i32x4;

#if defined(__has_builtin)
#if __has_builtin(__builtin_amdgcn_tensor_load_to_lds)
#define USE_TDM 1
#endif
#endif

union Frag {
    v16bf v;
    uint4 u[2];
};

// ---------------------------------------------------------------------------
// 1) flat f32 -> bf16 convert (vectorized x4)
// ---------------------------------------------------------------------------
__global__ __launch_bounds__(256) void cvt_flat(const float* __restrict__ src,
                                                __bf16* __restrict__ dst, int n4) {
    int i = blockIdx.x * blockDim.x + threadIdx.x;
    if (i >= n4) return;
    float4 v = ((const float4*)src)[i];
    union { __bf16 h[4]; uint2 u; } o;
    o.h[0] = (__bf16)v.x; o.h[1] = (__bf16)v.y;
    o.h[2] = (__bf16)v.z; o.h[3] = (__bf16)v.w;
    ((uint2*)dst)[i] = o.u;
}

// ---------------------------------------------------------------------------
// 2) f32 [K x N] row-major  ->  bf16 [N x K] (transposed), 32x32 LDS tiles
// ---------------------------------------------------------------------------
__global__ __launch_bounds__(256) void transpose_cvt(const float* __restrict__ src,
                                                     __bf16* __restrict__ dst,
                                                     int K, int N) {
    __shared__ float tile[32][33];
    int bn = blockIdx.x * 32;          // N offset
    int bk = blockIdx.y * 32;          // K offset
    int tx = threadIdx.x & 31;
    int ty = threadIdx.x >> 5;         // 0..7
#pragma unroll
    for (int r = 0; r < 32; r += 8)
        tile[ty + r][tx] = src[(size_t)(bk + ty + r) * N + (bn + tx)];
    __syncthreads();
#pragma unroll
    for (int r = 0; r < 32; r += 8)
        dst[(size_t)(bn + ty + r) * K + (bk + tx)] = (__bf16)tile[tx][ty + r];
}

// ---------------------------------------------------------------------------
// 3) butterfly coefficient tables from theta [CAP x 512]
//    v1[i][n] = cos(theta[i][p & 511]);  v2[i][n] = +/- sin(theta[i][p & 511])
//    p = (m << i) + j, j = n >> (10-i), m = n & ((1<<(10-i))-1)
// ---------------------------------------------------------------------------
__global__ __launch_bounds__(256) void make_tables(const float* __restrict__ theta,
                                                   float* __restrict__ v1,
                                                   float* __restrict__ v2) {
    int idx = blockIdx.x * blockDim.x + threadIdx.x;
    if (idx >= CAP * DOUT) return;
    int i = idx >> 10;
    int n = idx & 1023;
    int shift = 10 - i;
    int j = n >> shift;
    int m = n & ((1 << shift) - 1);
    int p = (m << i) + j;                       // 0..1023
    float t = theta[i * 512 + (p & 511)];
    float s = sinf(t), c = cosf(t);
    v1[idx] = c;
    v2[idx] = (p < 512) ? s : -s;
}

// ---------------------------------------------------------------------------
// 4) bf16 WMMA GEMM:  C[M x N](f32) = A[M x K](bf16,row) * Bt[N x K](bf16,row)^T
//    Block tile 128x128, 8 waves (4 row x 2 col), wave tile 32x64:
//    2x4 v_wmma_f32_16x16x32_bf16 accumulators, 8 WMMAs per wave per 32-K step.
//    Data movement: Tensor Data Mover (tensor_load_to_lds) with double-buffered
//    LDS; one TDM descriptor per operand per k-step, padded LDS rows (80B).
// ---------------------------------------------------------------------------
#define TM 128
#define TN 128
#define TK 32
#define LDA 40   // padded bf16 stride (80B = 16 dwords data + 4 dwords pad)
#define LDB 40
#define ABUF (TM * LDA)            // bf16 elems
#define BBUF (TN * LDB)
#define SBUF (ABUF + BBUF)

#ifdef USE_TDM
// Build a D# for a 2D tile: 128 rows x 32 bf16, row stride k_stride elems,
// DMA'd into LDS with 4-dword padding after every 16 dwords (row stride 80B).
__device__ __forceinline__ void tdm_load_tile(const __bf16* gptr,
                                              unsigned int lds_addr,
                                              int k_stride) {
    unsigned long long ga = (unsigned long long)(uintptr_t)gptr;
    u32x4 g0;
    g0[0] = 1u;                                            // count=1 (valid user D#)
    g0[1] = lds_addr;                                      // lds_addr (bytes)
    g0[2] = (unsigned int)ga;                              // global_addr[31:0]
    g0[3] = ((unsigned int)(ga >> 32) & 0x01FFFFFFu)       // global_addr[56:32]
          | 0x80000000u;                                   // type=2 (bits 127:126 = 10b)
    i32x8 g1;
    g1[0] = (1 << 16)                                      // data_size = 2 bytes
          | (1 << 20)                                      // pad_enable
          | (3 << 22)                                      // pad_interval: 16 dwords
          | (3 << 25);                                     // pad_amount: 4 dwords
    g1[1] = (k_stride & 0xFFFF) << 16;                     // tensor_dim0[15:0]
    g1[2] = (((unsigned)k_stride >> 16) & 0xFFFF)          // tensor_dim0[31:16]
          | (128 << 16);                                   // tensor_dim1[15:0] = 128
    g1[3] = (TK << 16);                                    // tile_dim0 = 32 elems
    g1[4] = 128;                                           // tile_dim1 = 128 rows
    g1[5] = k_stride;                                      // tensor_dim0_stride[31:0]
    g1[6] = 0;                                             // stride hi / dim1_stride lo
    g1[7] = 0;
    i32x4 z4 = {0, 0, 0, 0};
    i32x8 z8 = {0, 0, 0, 0, 0, 0, 0, 0};
    // amdgpu-toolchain (clang-23) 6-arg form:
    // (g0, g1, g2, g3, extra, cpol)
    __builtin_amdgcn_tensor_load_to_lds(g0, g1, z4, z4, z8, 0);
}
#endif

__global__ __launch_bounds__(256) void gemm_bf16_wmma(
        const __bf16* __restrict__ A,   // [M x K]
        const __bf16* __restrict__ Bt,  // [N x K]
        float* __restrict__ C,          // [M x N]
        int M, int N, int K) {
    const int tid  = threadIdx.x;
    const int wave = tid >> 5;              // 0..7
    const int lane = tid & 31;
    const int bm = blockIdx.y * TM;
    const int bn = blockIdx.x * TN;
    const int wm = (wave & 3) * 32;         // wave M offset in block tile
    const int wn = (wave >> 2) * 64;        // wave N offset in block tile

    v8f acc[2][4];
#pragma unroll
    for (int i = 0; i < 2; ++i)
#pragma unroll
        for (int j = 0; j < 4; ++j)
#pragma unroll
            for (int r = 0; r < 8; ++r) acc[i][j][r] = 0.0f;

    const int ml = lane & 15;               // M (or N) within fragment
    const int ka = (lane >> 4) * 8;         // A: K sub-offset per lane half
    const int kb = (lane >> 4) * 8;         // B: K sub-offset per lane half

#ifdef USE_TDM
    // ---------------- TDM double-buffered pipeline ----------------
    __shared__ __attribute__((aligned(16))) __bf16 smem[2 * SBUF];
    const __bf16* Abase = A  + (size_t)bm * K;
    const __bf16* Bbase = Bt + (size_t)bn * K;
    const unsigned int ldsA0 = (unsigned int)(uintptr_t)&smem[0];
    const unsigned int ldsB0 = ldsA0 + ABUF * 2;

    if (tid < 32) {                          // wave 0 drives the TDM
        tdm_load_tile(Abase, ldsA0, K);
        tdm_load_tile(Bbase, ldsB0, K);
    }
    int p = 0;
    for (int k0 = 0; k0 < K; k0 += TK) {
        if (tid < 32) __builtin_amdgcn_s_wait_tensorcnt(0);
        __syncthreads();                     // publish buf p; retire reads of p^1
        if (tid < 32 && (k0 + TK) < K) {
            tdm_load_tile(Abase + k0 + TK, ldsA0 + (p ^ 1) * (SBUF * 2), K);
            tdm_load_tile(Bbase + k0 + TK, ldsB0 + (p ^ 1) * (SBUF * 2), K);
        }
        const __bf16* sAp = smem + p * SBUF;
        const __bf16* sBp = sAp + ABUF;

        Frag fa[2], fb[4];
#pragma unroll
        for (int i = 0; i < 2; ++i) {
            const __bf16* ap = &sAp[(size_t)(wm + i * 16 + ml) * LDA + ka];
            fa[i].u[0] = *(const uint4*)ap;          // K = ka .. ka+7
            fa[i].u[1] = *(const uint4*)(ap + 16);   // K = 16+ka .. 16+ka+7
        }
#pragma unroll
        for (int j = 0; j < 4; ++j) {
            const __bf16* bp = &sBp[(size_t)(wn + j * 16 + ml) * LDB + kb];
            fb[j].u[0] = *(const uint4*)bp;          // K = kb .. kb+7
            fb[j].u[1] = *(const uint4*)(bp + 16);   // K = 16+kb .. 16+kb+7
        }
#pragma unroll
        for (int i = 0; i < 2; ++i)
#pragma unroll
            for (int j = 0; j < 4; ++j)
                acc[i][j] = __builtin_amdgcn_wmma_f32_16x16x32_bf16(
                    false, fa[i].v, false, fb[j].v,
                    (short)0, acc[i][j], false, false);
        p ^= 1;
    }
#else
    // ---------------- fallback: VGPR-staged global->LDS ----------------
    __shared__ __attribute__((aligned(16))) __bf16 smem[SBUF];
    __bf16* sA = smem;
    __bf16* sB = smem + ABUF;
    const int lrow = tid >> 2;
    const int lcol = (tid & 3) * 8;
    for (int k0 = 0; k0 < K; k0 += TK) {
        const __bf16* Ab = A  + (size_t)bm * K + k0;
        const __bf16* Bb = Bt + (size_t)bn * K + k0;
        *(uint4*)&sA[(size_t)lrow * LDA + lcol] =
            *(const uint4*)&Ab[(size_t)lrow * K + lcol];
        *(uint4*)&sA[(size_t)(lrow + 64) * LDA + lcol] =
            *(const uint4*)&Ab[(size_t)(lrow + 64) * K + lcol];
        *(uint4*)&sB[(size_t)lrow * LDB + lcol] =
            *(const uint4*)&Bb[(size_t)lrow * K + lcol];
        *(uint4*)&sB[(size_t)(lrow + 64) * LDB + lcol] =
            *(const uint4*)&Bb[(size_t)(lrow + 64) * K + lcol];
        __syncthreads();

        Frag fa[2], fb[4];
#pragma unroll
        for (int i = 0; i < 2; ++i) {
            const __bf16* ap = &sA[(size_t)(wm + i * 16 + ml) * LDA + ka];
            fa[i].u[0] = *(const uint4*)ap;
            fa[i].u[1] = *(const uint4*)(ap + 16);
        }
#pragma unroll
        for (int j = 0; j < 4; ++j) {
            const __bf16* bp = &sB[(size_t)(wn + j * 16 + ml) * LDB + kb];
            fb[j].u[0] = *(const uint4*)bp;
            fb[j].u[1] = *(const uint4*)(bp + 16);
        }
#pragma unroll
        for (int i = 0; i < 2; ++i)
#pragma unroll
            for (int j = 0; j < 4; ++j)
                acc[i][j] = __builtin_amdgcn_wmma_f32_16x16x32_bf16(
                    false, fa[i].v, false, fb[j].v,
                    (short)0, acc[i][j], false, false);
        __syncthreads();
    }
#endif

    // ---- store C (f32 16x16 D layout: reg r -> M = mh + r, lane&15 -> N) ----
    const int mh = (lane >> 4) * 8;
#pragma unroll
    for (int i = 0; i < 2; ++i)
#pragma unroll
        for (int j = 0; j < 4; ++j) {
            float* cp = C + (size_t)(bm + wm + i * 16 + mh) * N
                          + (bn + wn + j * 16 + ml);
#pragma unroll
            for (int r = 0; r < 8; ++r) cp[(size_t)r * N] = acc[i][j][r];
        }
}

// ---------------------------------------------------------------------------
// 5) fused butterfly + gates + modReLU + blend. One block = one batch row.
// ---------------------------------------------------------------------------
__global__ __launch_bounds__(256) void fused_tail(
        const float* __restrict__ state,
        const float* __restrict__ Ux,      // [B x 3072]  (cx | rx | gx)
        const float* __restrict__ Rlin,    // [B x 1024]  state @ W_r
        const float* __restrict__ Glin,    // [B x 1024]  state @ W_g
        const float* __restrict__ v1,      // [CAP x 1024]
        const float* __restrict__ v2,
        const float* __restrict__ br,
        const float* __restrict__ bg,
        const float* __restrict__ bc,
        float* __restrict__ out) {
    __shared__ float h[2][DOUT];
    const int row = blockIdx.x;
    const int tid = threadIdx.x;
    const float* srow = state + (size_t)row * DOUT;

#pragma unroll
    for (int e = 0; e < 4; ++e) {
        int n = tid + e * 256;
        h[0][n] = srow[n];
    }
    __syncthreads();

    int cur = 0;
#pragma unroll
    for (int i = 0; i < CAP; ++i) {
        const int mask = 512 >> i;
        const float* V1 = v1 + i * DOUT;
        const float* V2 = v2 + i * DOUT;
#pragma unroll
        for (int e = 0; e < 4; ++e) {
            int n = tid + e * 256;
            int p = n ^ mask;
            h[cur ^ 1][n] = h[cur][n] * V1[n] + h[cur][p] * V2[p];
        }
        cur ^= 1;
        __syncthreads();
    }

    const float* uc = Ux + (size_t)row * 3072;
    const float* ur = uc + 1024;
    const float* ug = uc + 2048;
    const float* rl = Rlin + (size_t)row * DOUT;
    const float* gl = Glin + (size_t)row * DOUT;
    float* orow = out + (size_t)row * DOUT;

#pragma unroll
    for (int e = 0; e < 4; ++e) {
        int n = tid + e * 256;
        float s = srow[n];
        float r = 1.0f / (1.0f + expf(-(ur[n] + rl[n] + br[n])));
        float g = 1.0f / (1.0f + expf(-(ug[n] + gl[n] + bg[n])));
        float pre = r * h[cur][n] + uc[n];
        float a = fabsf(pre) + 0.001f + bc[n];
        a = fmaxf(a, 0.0f);                       // relu
        float c = (pre > 0.0f) ? a : ((pre < 0.0f) ? -a : 0.0f);
        orow[n] = g * s + (1.0f - g) * c;
    }
}

// ---------------------------------------------------------------------------
// launch
// ---------------------------------------------------------------------------
extern "C" void kernel_launch(void* const* d_in, const int* in_sizes, int n_in,
                              void* d_out, int out_size, void* d_ws, size_t ws_size,
                              hipStream_t stream) {
    const float* x     = (const float*)d_in[0];
    const float* state = (const float*)d_in[1];
    const float* theta = (const float*)d_in[2];
    const float* U     = (const float*)d_in[3];
    const float* W_r   = (const float*)d_in[4];
    const float* W_g   = (const float*)d_in[5];
    const float* b_r   = (const float*)d_in[6];
    const float* b_g   = (const float*)d_in[7];
    const float* b_c   = (const float*)d_in[8];
    float* out = (float*)d_out;

    char* ws = (char*)d_ws;
    size_t off = 0;
    auto alloc = [&](size_t bytes) -> void* {
        void* p = ws + off;
        off += (bytes + 255) & ~(size_t)255;
        return p;
    };

    __bf16* xb  = (__bf16*)alloc((size_t)BATCH * DIN * 2);      // 16 MB
    __bf16* sb  = (__bf16*)alloc((size_t)BATCH * DOUT * 2);     // 16 MB
    __bf16* Utb = (__bf16*)alloc((size_t)3 * DOUT * DIN * 2);   //  6 MB  [3D x K]
    __bf16* Wrt = (__bf16*)alloc((size_t)DOUT * DOUT * 2);      //  2 MB
    __bf16* Wgt = (__bf16*)alloc((size_t)DOUT * DOUT * 2);      //  2 MB
    float*  Ux  = (float*) alloc((size_t)BATCH * 3 * DOUT * 4); // 96 MB
    float*  Rl  = (float*) alloc((size_t)BATCH * DOUT * 4);     // 32 MB
    float*  Gl  = (float*) alloc((size_t)BATCH * DOUT * 4);     // 32 MB
    float*  v1  = (float*) alloc((size_t)CAP * DOUT * 4);
    float*  v2  = (float*) alloc((size_t)CAP * DOUT * 4);

    // convert activations to bf16
    {
        int n4 = BATCH * DIN / 4;
        cvt_flat<<<(n4 + 255) / 256, 256, 0, stream>>>(x, xb, n4);
        cvt_flat<<<(n4 + 255) / 256, 256, 0, stream>>>(state, sb, n4);
    }
    // transpose-convert weights:  [K x N] f32 -> [N x K] bf16
    transpose_cvt<<<dim3(3 * DOUT / 32, DIN / 32), 256, 0, stream>>>(U,   Utb, DIN,  3 * DOUT);
    transpose_cvt<<<dim3(DOUT / 32,     DOUT / 32), 256, 0, stream>>>(W_r, Wrt, DOUT, DOUT);
    transpose_cvt<<<dim3(DOUT / 32,     DOUT / 32), 256, 0, stream>>>(W_g, Wgt, DOUT, DOUT);
    // butterfly coefficient tables
    make_tables<<<(CAP * DOUT + 255) / 256, 256, 0, stream>>>(theta, v1, v2);

    // GEMMs (all K = 1024)
    gemm_bf16_wmma<<<dim3(3 * DOUT / TN, BATCH / TM), 256, 0, stream>>>(
        xb, Utb, Ux, BATCH, 3 * DOUT, DIN);
    gemm_bf16_wmma<<<dim3(DOUT / TN, BATCH / TM), 256, 0, stream>>>(
        sb, Wrt, Rl, BATCH, DOUT, DOUT);
    gemm_bf16_wmma<<<dim3(DOUT / TN, BATCH / TM), 256, 0, stream>>>(
        sb, Wgt, Gl, BATCH, DOUT, DOUT);

    // fused butterfly + gates + modReLU
    fused_tail<<<BATCH, 256, 0, stream>>>(state, Ux, Rl, Gl, v1, v2,
                                          b_r, b_g, b_c, out);
}
